// RankActivation_32212254720543
// MI455X (gfx1250) — matrix operand order repaired and checked by reading, not compiled, not run on hardware
//
#include <hip/hip_runtime.h>
#include <hip/hip_bf16.h>
#include <math.h>

// Problem constants (from reference): B=32768, D=512, F=512, fp32 in/out.
#define B_SZ 32768
#define D_SZ 512
#define F_SZ 512
#define KSTEPS (D_SZ / 32)   // 16 K-steps of 32

typedef __bf16 v16bf __attribute__((ext_vector_type(16)));
typedef __bf16 v4bf  __attribute__((ext_vector_type(4)));
typedef float  v8f   __attribute__((ext_vector_type(8)));
typedef float  v4f   __attribute__((ext_vector_type(4)));

// ---------------------------------------------------------------------------
// One-time W -> (Wh, Wl) bf16 hi/lo split into scratch. W is 1 MB, L2-resident,
// so this removes per-row-block re-conversion (2048x) of W entirely.
// ---------------------------------------------------------------------------
__global__ __launch_bounds__(256) void split_w_bf16(const float* __restrict__ W,
                                                    __bf16* __restrict__ Wh,
                                                    __bf16* __restrict__ Wl) {
  const int i = (blockIdx.x * 256 + threadIdx.x) * 4;
  if (i < F_SZ * D_SZ) {
    v4f w = *(const v4f*)(W + i);
    v4bf h, l;
#pragma unroll
    for (int e = 0; e < 4; ++e) {
      __bf16 hh = (__bf16)w[e];
      h[e] = hh;
      l[e] = (__bf16)(w[e] - (float)hh);
    }
    *(v4bf*)(Wh + i) = h;
    *(v4bf*)(Wl + i) = l;
  }
}

// ---------------------------------------------------------------------------
// Main GEMM + tanh epilogue.
//
// Block: 256 threads = 8 waves. Block owns 16 output rows (one x slab),
// wave w owns columns [w*64, w*64+64) as 4 WMMA N-subtiles.
//
// Phase 1: cooperatively load the 16x512 fp32 x-slab ONCE, bf16 hi/lo split it,
//          and store it to LDS pre-swizzled into per-(Kstep,lane) 16-element
//          WMMA A-fragments (16 KB hi + 16 KB lo).
// Phase 2: per K-step, each wave grabs A via ds_load (no conversion VALU),
//          issues all 4 N-tiles' B loads as one clause, then runs the
//          bf16x3 WMMA chain: acc += Ah*Bh + Al*Bh + Ah*Bl (fp32 accum,
//          dropped Al*Bl term is O(2^-16) relative -> ~fp32 accuracy).
// ---------------------------------------------------------------------------
template <bool PRE>
__global__ __launch_bounds__(256) void rank_act_wmma(
    const float* __restrict__ x, const float* __restrict__ W,
    const __bf16* __restrict__ Wh, const __bf16* __restrict__ Wl,
    const float* __restrict__ avec, const float* __restrict__ bvec,
    float* __restrict__ out) {
  // Pre-swizzled A fragments: [kstep][lane][16 elems], bf16 hi and lo.
  __shared__ __align__(32) __bf16 ldsAh[KSTEPS * 32 * 16];
  __shared__ __align__(32) __bf16 ldsAl[KSTEPS * 32 * 16];

  const int tid      = threadIdx.x;
  const int lane     = tid & 31;           // wave32 on gfx1250
  const int wave     = tid >> 5;           // 8 waves per block
  const int laneMod  = lane & 15;
  const int laneHalf = lane >> 4;

  const int r0 = blockIdx.x * 16;          // 16 output rows per block (shared A)
  const int c0 = wave * 64;                // 64 output cols per wave (4 subtiles)

  // ---------------- Phase 1: stage x slab into LDS as WMMA fragments --------
  // 512 (kstep,lane) fragments, 256 threads -> 2 fragments per thread.
  // Fragment (s, l): row = l&15; laneHalf=l>>4 selects K {s*32+lh*16..+15}
  // (contiguous order, matching the B fragments).
#pragma unroll
  for (int p = 0; p < 2; ++p) {
    const int idx = tid + p * 256;         // 0..511
    const int s   = idx >> 5;              // kstep
    const int l   = idx & 31;              // lane the fragment belongs to
    const int lm  = l & 15;
    const int lh  = l >> 4;
    const float* xp = x + (size_t)(r0 + lm) * D_SZ + s * 32 + lh * 16;
    v4f f0 = *(const v4f*)(xp + 0);
    v4f f1 = *(const v4f*)(xp + 4);
    v4f f2 = *(const v4f*)(xp + 8);
    v4f f3 = *(const v4f*)(xp + 12);
    float afv[16];
#pragma unroll
    for (int e = 0; e < 4; ++e) {
      afv[e]      = f0[e];
      afv[4 + e]  = f1[e];
      afv[8 + e]  = f2[e];
      afv[12 + e] = f3[e];
    }
    v16bf h, lo;
#pragma unroll
    for (int e = 0; e < 16; ++e) {
      __bf16 hh = (__bf16)afv[e];
      h[e]  = hh;
      lo[e] = (__bf16)(afv[e] - (float)hh);
    }
    *(v16bf*)(ldsAh + idx * 16) = h;
    *(v16bf*)(ldsAl + idx * 16) = lo;
  }
  __syncthreads();

  // ---------------- Phase 2: K loop with WMMA ------------------------------
  v8f acc[4];
#pragma unroll
  for (int n = 0; n < 4; ++n)
    acc[n] = (v8f){0.f, 0.f, 0.f, 0.f, 0.f, 0.f, 0.f, 0.f};

  for (int s = 0; s < KSTEPS; ++s) {
    const int k = s * 32;

    // A fragments from LDS (no conversion work in the hot loop).
    v16bf ah = *(const v16bf*)(ldsAh + (s * 32 + lane) * 16);
    v16bf al = *(const v16bf*)(ldsAl + (s * 32 + lane) * 16);

    // Issue ALL 4 N-tiles' B loads up front (independent registers -> one
    // big load clause, WMMAs below retire against staged waits).
    v16bf bh[4], bl[4];
#pragma unroll
    for (int n = 0; n < 4; ++n) {
      const int f = c0 + n * 16 + laneMod;   // B 32x16: lane = column f,
      if (PRE) {                             // 16 contiguous K per lane.
        const size_t off = (size_t)f * D_SZ + k + laneHalf * 16;
        bh[n] = *(const v16bf*)(Wh + off);
        bl[n] = *(const v16bf*)(Wl + off);
      } else {
        const float* wrow = W + (size_t)f * D_SZ + k + laneHalf * 16;
#pragma unroll
        for (int q = 0; q < 4; ++q) {
          v4f wv = *(const v4f*)(wrow + q * 4);
#pragma unroll
          for (int e = 0; e < 4; ++e) {
            __bf16 hh = (__bf16)wv[e];
            bh[n][4 * q + e] = hh;
            bl[n][4 * q + e] = (__bf16)(wv[e] - (float)hh);
          }
        }
      }
    }

#pragma unroll
    for (int n = 0; n < 4; ++n) {
      acc[n] = __builtin_amdgcn_wmma_f32_16x16x32_bf16(
          false, ah, false, bh[n], (short)0, acc[n], false, false);
      acc[n] = __builtin_amdgcn_wmma_f32_16x16x32_bf16(
          false, al, false, bh[n], (short)0, acc[n], false, false);
      acc[n] = __builtin_amdgcn_wmma_f32_16x16x32_bf16(
          false, ah, false, bl[n], (short)0, acc[n], false, false);
    }
  }

  // ---------------- Epilogue: out = tanh(a*z + b) --------------------------
  // C/D layout: VGPR r -> M = r + 8*laneHalf, N = laneMod.
#pragma unroll
  for (int n = 0; n < 4; ++n) {
    const int f = c0 + n * 16 + laneMod;
    const float sa = avec[f];
    const float sb = bvec[f];
#pragma unroll
    for (int r = 0; r < 8; ++r) {
      const int row = r0 + laneHalf * 8 + r;
      out[(size_t)row * F_SZ + f] = tanhf(sa * acc[n][r] + sb);
    }
  }
}

extern "C" void kernel_launch(void* const* d_in, const int* in_sizes, int n_in,
                              void* d_out, int out_size, void* d_ws, size_t ws_size,
                              hipStream_t stream) {
  const float* x = (const float*)d_in[0];
  const float* W = (const float*)d_in[1];
  const float* a = (const float*)d_in[2];
  const float* b = (const float*)d_in[3];
  float* out = (float*)d_out;

  const size_t welems = (size_t)F_SZ * D_SZ;
  const size_t need   = 2 * welems * sizeof(__bf16);  // Wh + Wl = 1 MB

  dim3 block(256);
  dim3 grid(B_SZ / 16);  // 2048 row-blocks x 8 waves x 4 subtiles = 32768 x 512

  if (ws_size >= need) {
    __bf16* Wh = (__bf16*)d_ws;
    __bf16* Wl = Wh + welems;
    split_w_bf16<<<(int)((welems / 4 + 255) / 256), 256, 0, stream>>>(W, Wh, Wl);
    rank_act_wmma<true><<<grid, block, 0, stream>>>(x, W, Wh, Wl, a, b, out);
  } else {
    rank_act_wmma<false><<<grid, block, 0, stream>>>(x, W, nullptr, nullptr, a, b, out);
  }
}
